// CosineSimAttention_19937238188629
// MI455X (gfx1250) — matrix operand order repaired and checked by reading, not compiled, not run on hardware
//
#include <hip/hip_runtime.h>
#include <hip/hip_bf16.h>

// CDNA5 / gfx1250: wave32, WMMA 16x16x32 bf16 -> f32
typedef __attribute__((ext_vector_type(16))) __bf16 v16bf;
typedef __attribute__((ext_vector_type(8)))  __bf16 v8bf;
typedef __attribute__((ext_vector_type(4)))  __bf16 v4bf;
typedef __attribute__((ext_vector_type(8)))  float  v8f;
typedef __attribute__((ext_vector_type(4)))  float  v4f;

#define WMMA_BF16(a, b, c) \
  __builtin_amdgcn_wmma_f32_16x16x32_bf16(false, (a), false, (b), (short)0, (c), false, false)

// Problem sizes
constexpr int Bn = 2, Sn = 2048, Dn = 1024, Hn = 16, DHn = 64;
constexpr int Mrows = Bn * Sn;                       // 4096
constexpr size_t HEADSZ = (size_t)Sn * DHn;          // 131072 elems per (b,h)
constexpr size_t QKVSZ  = (size_t)Bn * Hn * HEADSZ;  // 4194304 elems per tensor

constexpr int LPAD = 40;              // padded K-stride (bf16 elems) of LDS B-tile
constexpr int LBUF = 64 * LPAD;       // 2560 elems (5 KB) per buffer

// ---------------------------------------------------------------------------
// Cooperative staging of a 32(K) x 64(N) fp32 weight tile into LDS as bf16,
// transposed to [N][K] (stride LPAD) so B-fragments are contiguous in K.
// 128 threads: thread -> 4x4 sub-block, coalesced v4f global loads.
// ---------------------------------------------------------------------------
__device__ inline void stage_wtile(const float* __restrict__ W, int ldw,
                                   int ncol0, int ks, __bf16* lds, int tid) {
  const int kg = tid >> 4;   // 0..7  -> K group of 4
  const int ng = tid & 15;   // 0..15 -> N group of 4
  const float* src = W + (size_t)(ks + kg * 4) * ldw + ncol0 + ng * 4;
  v4f r0 = *(const v4f*)(src);
  v4f r1 = *(const v4f*)(src + ldw);
  v4f r2 = *(const v4f*)(src + 2 * (size_t)ldw);
  v4f r3 = *(const v4f*)(src + 3 * (size_t)ldw);
#pragma unroll
  for (int nn = 0; nn < 4; ++nn) {
    v4bf t;
    t[0] = (__bf16)r0[nn]; t[1] = (__bf16)r1[nn];
    t[2] = (__bf16)r2[nn]; t[3] = (__bf16)r3[nn];
    *(v4bf*)(lds + (ng * 4 + nn) * LPAD + kg * 4) = t;   // 8-byte ds_store
  }
}

// Read B fragment j from staged LDS tile: lane col = j*16+ln, K = (i&7) + 16*(i>>3) + 8*hi.
__device__ inline v16bf ldsb_frag(const __bf16* lds, int j, int ln, int hi) {
  const __bf16* p = lds + (j * 16 + ln) * LPAD + hi * 8;
  v8bf lo = *(const v8bf*)p;        // ds_load_b128
  v8bf h8 = *(const v8bf*)(p + 16);  // ds_load_b128
  v16bf out;
#pragma unroll
  for (int i = 0; i < 8; ++i) { out[i] = lo[i]; out[8 + i] = h8[i]; }
  return out;
}

// ---------------------------------------------------------------------------
// Kernel 1: fused QKV projection + per-head L2 normalization (+ exp(scale) on Q)
// Tiles 64(M) x 64(N); 4 waves, wave = 16 rows x 64 cols.
// Q,K stored [B,H,S,DH]; V stored transposed [B,H,DH,S] for the P@V GEMM.
// ---------------------------------------------------------------------------
__global__ __launch_bounds__(128) void qkv_kernel(
    const float* __restrict__ x, const float* __restrict__ w_q,
    const float* __restrict__ w_kv, const float* __restrict__ scale,
    __bf16* __restrict__ Qo, __bf16* __restrict__ Ko, __bf16* __restrict__ Vt)
{
  __shared__ __align__(16) __bf16 smem[2 * LBUF];
  const int tid = threadIdx.x;
  const int lane = tid & 31, wave = tid >> 5;
  const int hi = lane >> 4, ln = lane & 15;
  const int m0 = blockIdx.x * 64 + wave * 16;
  const int ntile = blockIdx.y;                 // 0-15 Q, 16-31 K, 32-47 V

  const float* W; int ldw, ncol0;
  if (ntile < 16) { W = w_q;  ldw = 1024; ncol0 = ntile * 64; }
  else            { W = w_kv; ldw = 2048; ncol0 = ntile * 64 - 1024; }

  v8f acc[4] = {};

  stage_wtile(W, ldw, ncol0, 0, smem, tid);
  __syncthreads();

  for (int ks = 0; ks < Dn; ks += 32) {
    __bf16* cur = smem + ((ks >> 5) & 1) * LBUF;
    if (ks + 32 < Dn)
      stage_wtile(W, ldw, ncol0, ks + 32, smem + (((ks >> 5) + 1) & 1) * LBUF, tid);

    // A fragment: row = m0+ln, K = ks + (i&7) + 16*(i>>3) + 8*hi
    const float* ax = x + (size_t)(m0 + ln) * Dn + ks + hi * 8;
    v4f a0 = *(const v4f*)(ax);
    v4f a1 = *(const v4f*)(ax + 4);
    v4f a2 = *(const v4f*)(ax + 16);
    v4f a3 = *(const v4f*)(ax + 20);
    v16bf afrag;
#pragma unroll
    for (int i = 0; i < 4; ++i) {
      afrag[i]      = (__bf16)a0[i];
      afrag[4 + i]  = (__bf16)a1[i];
      afrag[8 + i]  = (__bf16)a2[i];
      afrag[12 + i] = (__bf16)a3[i];
    }
#pragma unroll
    for (int j = 0; j < 4; ++j)
      acc[j] = WMMA_BF16(afrag, ldsb_frag(cur, j, ln, hi), acc[j]);
    __syncthreads();
  }

  // Epilogue: L2 norm over the head (64 cols) for Q/K tiles; Q *= exp(scale[h])
  const bool is_q = (ntile < 16);
  const bool is_k = (ntile >= 16 && ntile < 32);
  if (is_q || is_k) {
    const float smul = is_q ? __expf(scale[ntile]) : 1.0f;
#pragma unroll
    for (int r = 0; r < 8; ++r) {
      float p = 0.f;
#pragma unroll
      for (int j = 0; j < 4; ++j) p += acc[j][r] * acc[j][r];
#pragma unroll
      for (int mask = 1; mask <= 8; mask <<= 1) p += __shfl_xor(p, mask, 32);
      const float inv = smul / fmaxf(sqrtf(p), 1e-12f);
#pragma unroll
      for (int j = 0; j < 4; ++j) acc[j][r] *= inv;
    }
  }

  if (!is_q && !is_k) {
    // ---- V: transpose 16x64 wave tile through LDS, store [B,H,DH,S] ----
    const int h = ntile - 32;
    __bf16* tp = smem + wave * 1024;            // 16 rows x 64 dh, row-major
#pragma unroll
    for (int r = 0; r < 8; ++r)
#pragma unroll
      for (int j = 0; j < 4; ++j)
        tp[(r + 8 * hi) * 64 + j * 16 + ln] = (__bf16)acc[j][r];
    __builtin_amdgcn_fence(__ATOMIC_ACQ_REL, "wavefront");

    const int b = m0 >> 11, s0 = m0 & (Sn - 1);
#pragma unroll
    for (int c = 0; c < 2; ++c) {
      const int dh = lane + c * 32;
      v16bf col;
#pragma unroll
      for (int i = 0; i < 16; ++i) col[i] = tp[i * 64 + dh];
      *(v16bf*)(Vt + ((size_t)(b * Hn + h) * DHn + dh) * Sn + s0) = col;
    }
    return;
  }

  __bf16* dst = is_q ? Qo : Ko;
  const int h = is_q ? ntile : ntile - 16;
#pragma unroll
  for (int r = 0; r < 8; ++r) {
    const int m = m0 + r + 8 * hi;
    const int b = m >> 11, s = m & (Sn - 1);
    __bf16* rowp = dst + ((size_t)(b * Hn + h) * Sn + s) * DHn;
#pragma unroll
    for (int j = 0; j < 4; ++j) rowp[j * 16 + ln] = (__bf16)acc[j][r];
  }
}

// ---------------------------------------------------------------------------
// Kernel 2: flash attention per (b, h, 64-query tile). 4 waves, 16 q-rows each.
// K in [B,H,S,DH]; V transposed [B,H,DH,S] -> all fragments contiguous b128.
// P staging is per-wave LDS: same-wave LDS ops complete in order (ISA 7.3),
// so no block barrier is needed -- only a wavefront-scope compiler fence.
// ---------------------------------------------------------------------------
__global__ __launch_bounds__(128) void attn_kernel(
    const __bf16* __restrict__ Q, const __bf16* __restrict__ K,
    const __bf16* __restrict__ Vt, __bf16* __restrict__ O)
{
  __shared__ __align__(32) __bf16 pbuf[4][16 * 64];   // per-wave P staging
  const int lane = threadIdx.x & 31;
  const int wave = threadIdx.x >> 5;
  const int hi = lane >> 4, ln = lane & 15;
  const int bh = blockIdx.z * Hn + blockIdx.y;
  const size_t base = (size_t)bh * HEADSZ;
  const int q0 = blockIdx.x * 64 + wave * 16;
  __bf16* pw = &pbuf[wave][0];

  // Q fragments for the whole tile (DH=64 -> two K-steps of 32)
  v16bf qf[2];
#pragma unroll
  for (int ks = 0; ks < 2; ++ks) {
    const __bf16* qp = Q + base + (size_t)(q0 + ln) * DHn + ks * 32 + hi * 16;
    v8bf lo = *(const v8bf*)qp;
    v8bf h8 = *(const v8bf*)(qp + 8);
#pragma unroll
    for (int i = 0; i < 8; ++i) { qf[ks][i] = lo[i]; qf[ks][8 + i] = h8[i]; }
  }

  v8f oacc[4] = {};
  float mrun[8], lrun[8];
#pragma unroll
  for (int r = 0; r < 8; ++r) { mrun[r] = -1e30f; lrun[r] = 0.f; }

  for (int k0 = 0; k0 < Sn; k0 += 64) {
    // Prefetch next key/value tiles into cache while we compute this one.
    if (k0 + 64 < Sn) {
      __builtin_prefetch(K + base + (size_t)(k0 + 64 + lane) * DHn, 0, 1);
      __builtin_prefetch(Vt + base + (size_t)lane * Sn + k0 + 64, 0, 1);
    }

    // S = Q @ K^T   (16 q-rows x 64 keys)
    v8f sacc[4] = {};
#pragma unroll
    for (int ks = 0; ks < 2; ++ks) {
#pragma unroll
      for (int j = 0; j < 4; ++j) {
        // B frag: key = k0 + j*16 + ln (per lane), kdim = ks*32 + 16*hi + i
        const __bf16* kp = K + base + (size_t)(k0 + j * 16 + ln) * DHn + ks * 32 + hi * 16;
        v16bf bfrag = *(const v16bf*)kp;
        sacc[j] = WMMA_BF16(qf[ks], bfrag, sacc[j]);
      }
    }

    // online softmax (row r + 8*hi lives in VGPR slot r; reduce 16-lane group)
    float corr[8];
#pragma unroll
    for (int r = 0; r < 8; ++r) {
      float pm = fmaxf(fmaxf(sacc[0][r], sacc[1][r]), fmaxf(sacc[2][r], sacc[3][r]));
#pragma unroll
      for (int mask = 1; mask <= 8; mask <<= 1) pm = fmaxf(pm, __shfl_xor(pm, mask, 32));
      const float nm = fmaxf(mrun[r], pm);
      corr[r] = __expf(mrun[r] - nm);
      mrun[r] = nm;
      float ps = 0.f;
#pragma unroll
      for (int j = 0; j < 4; ++j) {
        const float e = __expf(sacc[j][r] - nm);
        sacc[j][r] = e;
        ps += e;
      }
#pragma unroll
      for (int mask = 1; mask <= 8; mask <<= 1) ps += __shfl_xor(ps, mask, 32);
      lrun[r] = lrun[r] * corr[r] + ps;
#pragma unroll
      for (int j = 0; j < 4; ++j) oacc[j][r] *= corr[r];
    }

    // stage P (C-layout) to per-wave LDS row-major [16][64] as bf16
#pragma unroll
    for (int r = 0; r < 8; ++r) {
      const int row = r + 8 * hi;
#pragma unroll
      for (int j = 0; j < 4; ++j) pw[row * 64 + j * 16 + ln] = (__bf16)sacc[j][r];
    }
    __builtin_amdgcn_fence(__ATOMIC_ACQ_REL, "wavefront");

    // O += P @ V  (V^T layout -> contiguous B fragments)
#pragma unroll
    for (int ks = 0; ks < 2; ++ks) {
      const __bf16* pp = pw + ln * 64 + ks * 32 + hi * 16;
      v8bf lo = *(const v8bf*)pp;
      v8bf h8 = *(const v8bf*)(pp + 8);
      v16bf pf;
#pragma unroll
      for (int i = 0; i < 8; ++i) { pf[i] = lo[i]; pf[8 + i] = h8[i]; }
#pragma unroll
      for (int j = 0; j < 4; ++j) {
        // B frag: col dh = j*16+ln (per lane), kdim(key) = k0+ks*32+16*hi+i
        const __bf16* vp = Vt + base + (size_t)(j * 16 + ln) * Sn + k0 + ks * 32 + hi * 16;
        v16bf bfrag = *(const v16bf*)vp;
        oacc[j] = WMMA_BF16(pf, bfrag, oacc[j]);
      }
    }
    __builtin_amdgcn_fence(__ATOMIC_ACQ_REL, "wavefront");
  }

  // finalize: O /= l, store bf16 [B,H,S,DH]
#pragma unroll
  for (int r = 0; r < 8; ++r) {
    const float inv = 1.0f / lrun[r];
    const int s = q0 + r + 8 * hi;
    __bf16* op = O + base + (size_t)s * DHn;
#pragma unroll
    for (int j = 0; j < 4; ++j) op[j * 16 + ln] = (__bf16)(oacc[j][r] * inv);
  }
}

// ---------------------------------------------------------------------------
// Kernel 3: out = attn_out @ w_out + b_out   (A in [B,H,S,DH] bf16 layout)
// ---------------------------------------------------------------------------
__global__ __launch_bounds__(128) void outproj_kernel(
    const __bf16* __restrict__ A, const float* __restrict__ Wo,
    const float* __restrict__ bias, float* __restrict__ out)
{
  __shared__ __align__(16) __bf16 smem[2 * LBUF];
  const int tid = threadIdx.x;
  const int lane = tid & 31, wave = tid >> 5;
  const int hi = lane >> 4, ln = lane & 15;
  const int m0 = blockIdx.x * 64 + wave * 16;
  const int n0 = blockIdx.y * 64;

  const int m = m0 + ln;                       // this lane's A row
  const int b = m >> 11, s = m & (Sn - 1);

  v8f acc[4] = {};
  stage_wtile(Wo, Dn, n0, 0, smem, tid);
  __syncthreads();

  for (int ks = 0; ks < 1024; ks += 32) {
    __bf16* cur = smem + ((ks >> 5) & 1) * LBUF;
    if (ks + 32 < 1024)
      stage_wtile(Wo, Dn, n0, ks + 32, smem + (((ks >> 5) + 1) & 1) * LBUF, tid);

    v16bf afrag;
#pragma unroll
    for (int half = 0; half < 2; ++half) {
      const int kn = ks + 8 * hi + half * 16;  // inner column (h*64 + dh)
      const int hh = kn >> 6, dh = kn & 63;
      const __bf16* ap = A + ((size_t)(b * Hn + hh) * Sn + s) * DHn + dh;
      v8bf t = *(const v8bf*)ap;
#pragma unroll
      for (int i = 0; i < 8; ++i) afrag[half * 8 + i] = t[i];
    }
#pragma unroll
    for (int j = 0; j < 4; ++j)
      acc[j] = WMMA_BF16(afrag, ldsb_frag(cur, j, ln, hi), acc[j]);
    __syncthreads();
  }

#pragma unroll
  for (int r = 0; r < 8; ++r) {
    const int mr = m0 + r + 8 * hi;
    float* op = out + (size_t)mr * Dn + n0;
#pragma unroll
    for (int j = 0; j < 4; ++j) {
      const int n = j * 16 + ln;
      op[n] = acc[j][r] + bias[n0 + n];
    }
  }
}

// ---------------------------------------------------------------------------
extern "C" void kernel_launch(void* const* d_in, const int* in_sizes, int n_in,
                              void* d_out, int out_size, void* d_ws, size_t ws_size,
                              hipStream_t stream) {
  const float* x     = (const float*)d_in[0];
  const float* w_q   = (const float*)d_in[1];
  const float* w_kv  = (const float*)d_in[2];
  const float* w_out = (const float*)d_in[3];
  const float* b_out = (const float*)d_in[4];
  const float* scale = (const float*)d_in[5];

  __bf16* Q  = (__bf16*)d_ws;          // [B,H,S,DH] bf16
  __bf16* K  = Q + QKVSZ;              // [B,H,S,DH]
  __bf16* Vt = K + QKVSZ;              // [B,H,DH,S]  (transposed)
  __bf16* AO = Vt + QKVSZ;             // [B,H,S,DH]

  qkv_kernel<<<dim3(Mrows / 64, 48), 128, 0, stream>>>(x, w_q, w_kv, scale, Q, K, Vt);
  attn_kernel<<<dim3(Sn / 64, Hn, Bn), 128, 0, stream>>>(Q, K, Vt, AO);
  outproj_kernel<<<dim3(Mrows / 64, Dn / 64), 128, 0, stream>>>(AO, w_out, b_out, (float*)d_out);
}
